// MultiHeadAttention_58265526338049
// MI455X (gfx1250) — matrix operand order repaired and checked by reading, not compiled, not run on hardware
//
#include <hip/hip_runtime.h>

// ---- problem constants (from reference setup_inputs) ----
#define B_   2
#define S_   2048
#define D_   1024
#define H_   16
#define HD_  64
#define SCALE_ 0.03125f   // D^-0.5 = 1024^-0.5 (reference scales by embedding dim)

typedef __attribute__((ext_vector_type(16))) __bf16       v16bf;
typedef __attribute__((ext_vector_type(8)))  float        v8f;
typedef __attribute__((ext_vector_type(4)))  unsigned int v4u;

union Frag16 { v16bf v; v4u u[2]; };

__device__ __forceinline__ int lane_id() { return (int)(threadIdx.x & 31u); }

// A-matrix 16x32 bf16 fragment from row-major [rows, ld] storage.
// Layout: lane<16 row=lane, elems 0..15 -> K=k0..k0+15
//         lane>=16 row=lane-16, elems 0..15 -> K=k0+16..k0+31
__device__ __forceinline__ v16bf load_a_frag(const __bf16* base, int ld, int row0, int k0) {
  int lane = lane_id(), half = lane >> 4, l = lane & 15;
  const __bf16* p = base + (long)(row0 + l) * ld + (k0 + half * 16);
  Frag16 f;
  f.u[0] = *(const v4u*)(p);
  f.u[1] = *(const v4u*)(p + 8);
  return f.v;
}

// B-matrix 32x16 bf16 fragment, from N-major storage baseN[n][k] (leading dim ld over K).
// Contiguous-k layout (same as A): lanes 0-15 col=lane hold K=k0..k0+15; lanes 16-31 hold K=k0+16..k0+31.
__device__ __forceinline__ v16bf load_b_frag(const __bf16* baseN, int ld, int n0, int k0) {
  int lane = lane_id(), half = lane >> 4, l = lane & 15;
  const __bf16* p = baseN + (long)(n0 + l) * ld + (k0 + half * 16);
  Frag16 f;
  f.u[0] = *(const v4u*)(p);
  f.u[1] = *(const v4u*)(p + 8);
  return f.v;
}

__device__ __forceinline__ v8f wmma_bf16(v16bf a, v16bf b, v8f c) {
  // (neg_a, A, neg_b, B, c_mod, C, reuse_a, reuse_b)
  return __builtin_amdgcn_wmma_f32_16x16x32_bf16(false, a, false, b, (short)0, c, false, false);
}

__device__ __forceinline__ v8f zero8() {
  v8f z = {0.f, 0.f, 0.f, 0.f, 0.f, 0.f, 0.f, 0.f};
  return z;
}

// ---------------- pack kernels ----------------
__global__ void cvt_bf16_kernel(const float* __restrict__ src, __bf16* __restrict__ dst, int n) {
  int i = blockIdx.x * blockDim.x + threadIdx.x;
  if (i < n) dst[i] = (__bf16)src[i];
}

// dst[batch][c][r] = src[batch][r][c]; src fp32 [batches][R][C], dst bf16 [batches][C][R]
__global__ void transpose_pack_kernel(const float* __restrict__ src, __bf16* __restrict__ dst,
                                      int R, int C) {
  long n = (long)R * C;
  long i = (long)blockIdx.x * blockDim.x + threadIdx.x;
  if (i >= n) return;
  int r = (int)(i / C), c = (int)(i % C);
  const float* s = src + (long)blockIdx.y * n;
  __bf16*      d = dst + (long)blockIdx.y * n;
  d[(long)c * R + r] = (__bf16)s[i];
}

// ---------------- fused QKV projection ----------------
// One wave: 16 seq rows x 64 head-dims of q,k,v for one (b,h).
// Loads for each K-step are issued as one batch before the 12 WMMAs so the
// compiler stages partial s_wait_loadcnt values instead of wait-0 per WMMA.
__global__ void qkv_kernel(const __bf16* __restrict__ xb,
                           const __bf16* __restrict__ wtq,
                           const __bf16* __restrict__ wtk,
                           const __bf16* __restrict__ wtv,
                           __bf16* __restrict__ qm,   // [B*H][S][HD]
                           __bf16* __restrict__ km,   // [B*H][S][HD]
                           __bf16* __restrict__ vt) { // [B*H][HD][S]
  const int S16 = S_ / 16;
  int w  = blockIdx.x * (blockDim.x >> 5) + (threadIdx.x >> 5);
  int st = w % S16;
  int bh = w / S16;
  int h  = bh % H_;
  int b  = bh / H_;
  int lane = lane_id(), half = lane >> 4, l = lane & 15;
  int r0 = st * 16;

  const __bf16* A  = xb  + (long)b * S_ * D_;
  const __bf16* Bq = wtq + (long)h * HD_ * D_;   // [HD][D] N-major
  const __bf16* Bk = wtk + (long)h * HD_ * D_;
  const __bf16* Bv = wtv + (long)h * HD_ * D_;

  v8f cq[4], ck[4], cv[4];
#pragma unroll
  for (int nt = 0; nt < 4; ++nt) { cq[nt] = zero8(); ck[nt] = zero8(); cv[nt] = zero8(); }

  for (int k0 = 0; k0 < D_; k0 += 32) {
    // issue all fragment loads for this K-step up front
    v16bf a = load_a_frag(A, D_, r0, k0);
    v16bf bq[4], bk[4], bv[4];
#pragma unroll
    for (int nt = 0; nt < 4; ++nt) {
      bq[nt] = load_b_frag(Bq, D_, nt * 16, k0);
      bk[nt] = load_b_frag(Bk, D_, nt * 16, k0);
      bv[nt] = load_b_frag(Bv, D_, nt * 16, k0);
    }
    // then run the 12 independent WMMAs
#pragma unroll
    for (int nt = 0; nt < 4; ++nt) {
      cq[nt] = wmma_bf16(a, bq[nt], cq[nt]);
      ck[nt] = wmma_bf16(a, bk[nt], ck[nt]);
      cv[nt] = wmma_bf16(a, bv[nt], cv[nt]);
    }
  }

  long qbase = (long)bh * S_ * HD_;
  long vbase = (long)bh * HD_ * S_;
#pragma unroll
  for (int nt = 0; nt < 4; ++nt) {
#pragma unroll
    for (int i = 0; i < 8; ++i) {
      int row = r0 + i + half * 8;        // seq index
      int col = nt * 16 + l;              // head-dim index
      qm[qbase + (long)row * HD_ + col] = (__bf16)cq[nt][i];
      km[qbase + (long)row * HD_ + col] = (__bf16)ck[nt][i];
      vt[vbase + (long)col * S_ + row]  = (__bf16)cv[nt][i];
    }
  }
}

// ---------------- flash attention (causal, online softmax) ----------------
#define WPB 4
__global__ void attn_kernel(const __bf16* __restrict__ qm,
                            const __bf16* __restrict__ km,
                            const __bf16* __restrict__ vt,
                            __bf16* __restrict__ ao) { // [B][S][D]
  __shared__ __align__(16) __bf16 pbuf[WPB][16 * 32];
  const int S16 = S_ / 16;
  int wib = threadIdx.x >> 5;
  int w   = blockIdx.x * WPB + wib;
  int qt  = w % S16;
  int bh  = w / S16;
  int h   = bh % H_;
  int b   = bh / H_;
  int lane = lane_id(), half = lane >> 4, l = lane & 15;
  int q0 = qt * 16;

  const __bf16* Q = qm + (long)bh * S_ * HD_;
  const __bf16* K = km + (long)bh * S_ * HD_;
  const __bf16* V = vt + (long)bh * HD_ * S_;   // [HD][S] N-major over hd

  v16bf aq0 = load_a_frag(Q, HD_, q0, 0);
  v16bf aq1 = load_a_frag(Q, HD_, q0, 32);

  float m[8], lsum[8];
  v8f co[4];
#pragma unroll
  for (int i = 0; i < 8; ++i) { m[i] = -__builtin_inff(); lsum[i] = 0.f; }
#pragma unroll
  for (int nt = 0; nt < 4; ++nt) co[nt] = zero8();

  __bf16* P = &pbuf[wib][0];

  for (int kb = 0; kb <= q0 + 15; kb += 32) {
    // issue ALL loads for this key block first: K-frags feed the score WMMAs,
    // V-frags are only needed after softmax, so their latency hides behind
    // the score WMMAs + the whole softmax VALU/shuffle/exp sequence.
    v16bf bk00 = load_b_frag(K, HD_, kb,      0);
    v16bf bk01 = load_b_frag(K, HD_, kb,      32);
    v16bf bk10 = load_b_frag(K, HD_, kb + 16, 0);
    v16bf bk11 = load_b_frag(K, HD_, kb + 16, 32);
    v16bf bv[4];
#pragma unroll
    for (int nt = 0; nt < 4; ++nt) bv[nt] = load_b_frag(V, S_, nt * 16, kb);

    // scores: Q(16x64) @ K^T(64x32) -> two 16x16 tiles
    v8f s0 = zero8(), s1 = zero8();
    s0 = wmma_bf16(aq0, bk00, s0);
    s0 = wmma_bf16(aq1, bk01, s0);
    s1 = wmma_bf16(aq0, bk10, s1);
    s1 = wmma_bf16(aq1, bk11, s1);

#pragma unroll
    for (int i = 0; i < 8; ++i) {
      int qi = q0 + i + half * 8;
      int kq0 = kb + l, kq1 = kb + 16 + l;
      s0[i] = (kq0 <= qi) ? s0[i] * SCALE_ : -__builtin_inff();
      s1[i] = (kq1 <= qi) ? s1[i] * SCALE_ : -__builtin_inff();

      // row max over the 32 keys of this block (reduce across 16-lane half)
      float bm = fmaxf(s0[i], s1[i]);
      bm = fmaxf(bm, __shfl_xor(bm, 1));
      bm = fmaxf(bm, __shfl_xor(bm, 2));
      bm = fmaxf(bm, __shfl_xor(bm, 4));
      bm = fmaxf(bm, __shfl_xor(bm, 8));
      float nm = fmaxf(m[i], bm);
      float f  = __expf(m[i] - nm);     // 0 on first block (m = -inf)
      m[i] = nm;
      lsum[i] *= f;
#pragma unroll
      for (int nt = 0; nt < 4; ++nt) co[nt][i] *= f;

      s0[i] = __expf(s0[i] - nm);       // masked -> exp(-inf) = 0
      s1[i] = __expf(s1[i] - nm);
      float rs = s0[i] + s1[i];
      rs += __shfl_xor(rs, 1);
      rs += __shfl_xor(rs, 2);
      rs += __shfl_xor(rs, 4);
      rs += __shfl_xor(rs, 8);
      lsum[i] += rs;

      // stash P tile (16x32 row-major bf16) in LDS to re-layout C-frag -> A-frag
      int row = i + half * 8;
      P[row * 32 + l]      = (__bf16)s0[i];
      P[row * 32 + 16 + l] = (__bf16)s1[i];
    }
    asm volatile("" ::: "memory");  // same-wave DS ops are HW-ordered; just stop reordering
    v16bf ap = load_a_frag(P, 32, 0, 0);
    asm volatile("" ::: "memory");

    // out += P(16x32) @ V(32x64); V-frags already in flight/registers
#pragma unroll
    for (int nt = 0; nt < 4; ++nt)
      co[nt] = wmma_bf16(ap, bv[nt], co[nt]);
  }

  // normalize and write concat-head output [B][S][D]
#pragma unroll
  for (int i = 0; i < 8; ++i) {
    float inv = 1.0f / lsum[i];
    int row = q0 + i + half * 8;
#pragma unroll
    for (int nt = 0; nt < 4; ++nt) {
      int col = h * HD_ + nt * 16 + l;
      ao[((long)b * S_ + row) * D_ + col] = (__bf16)(co[nt][i] * inv);
    }
  }
}

// ---------------- output projection: out = ao @ Wo + bo ----------------
__global__ void oproj_kernel(const __bf16* __restrict__ ao,   // [B*S][D] bf16
                             const __bf16* __restrict__ wot,  // [D][D] N-major (WoT[n][k])
                             const float* __restrict__ bo,
                             float* __restrict__ out) {       // [B*S][D] fp32
  const int NT64 = D_ / 64;
  int w    = blockIdx.x * (blockDim.x >> 5) + (threadIdx.x >> 5);
  int nb   = w % NT64;
  int rt   = w / NT64;
  int lane = lane_id(), half = lane >> 4, l = lane & 15;
  int r0 = rt * 16, n0 = nb * 64;

  v8f c[4];
#pragma unroll
  for (int j = 0; j < 4; ++j) c[j] = zero8();

  for (int k0 = 0; k0 < D_; k0 += 32) {
    v16bf a = load_a_frag(ao, D_, r0, k0);
    v16bf bw[4];
#pragma unroll
    for (int j = 0; j < 4; ++j) bw[j] = load_b_frag(wot, D_, n0 + j * 16, k0);
#pragma unroll
    for (int j = 0; j < 4; ++j) c[j] = wmma_bf16(a, bw[j], c[j]);
  }

#pragma unroll
  for (int j = 0; j < 4; ++j) {
#pragma unroll
    for (int i = 0; i < 8; ++i) {
      int row = r0 + i + half * 8;
      int n   = n0 + j * 16 + l;
      out[(long)row * D_ + n] = c[j][i] + bo[n];
    }
  }
}

extern "C" void kernel_launch(void* const* d_in, const int* in_sizes, int n_in,
                              void* d_out, int out_size, void* d_ws, size_t ws_size,
                              hipStream_t stream) {
  (void)in_sizes; (void)n_in; (void)out_size; (void)ws_size;
  const float* x  = (const float*)d_in[0];
  const float* Wq = (const float*)d_in[1];
  const float* Wk = (const float*)d_in[2];
  const float* Wv = (const float*)d_in[3];
  const float* Wo = (const float*)d_in[4];
  const float* bo = (const float*)d_in[5];
  float* out = (float*)d_out;

  const long NX  = (long)B_ * S_ * D_;   // 4,194,304
  const long NW  = (long)H_ * D_ * HD_;  // 1,048,576
  const long NWO = (long)D_ * D_;        // 1,048,576

  __bf16* ws  = (__bf16*)d_ws;
  __bf16* xb  = ws;  ws += NX;
  __bf16* wtq = ws;  ws += NW;
  __bf16* wtk = ws;  ws += NW;
  __bf16* wtv = ws;  ws += NW;
  __bf16* wot = ws;  ws += NWO;
  __bf16* qm  = ws;  ws += NX;
  __bf16* km  = ws;  ws += NX;
  __bf16* vt  = ws;  ws += NX;
  __bf16* ao  = ws;  ws += NX;

  cvt_bf16_kernel<<<(int)((NX + 255) / 256), 256, 0, stream>>>(x, xb, (int)NX);
  transpose_pack_kernel<<<dim3((D_ * HD_ + 255) / 256, H_), 256, 0, stream>>>(Wq, wtq, D_, HD_);
  transpose_pack_kernel<<<dim3((D_ * HD_ + 255) / 256, H_), 256, 0, stream>>>(Wk, wtk, D_, HD_);
  transpose_pack_kernel<<<dim3((D_ * HD_ + 255) / 256, H_), 256, 0, stream>>>(Wv, wtv, D_, HD_);
  transpose_pack_kernel<<<dim3((int)((NWO + 255) / 256), 1), 256, 0, stream>>>(Wo, wot, D_, D_);

  int waves_qkv = B_ * H_ * (S_ / 16);                 // 4096
  qkv_kernel<<<waves_qkv / 4, 128, 0, stream>>>(xb, wtq, wtk, wtv, qm, km, vt);
  attn_kernel<<<waves_qkv / WPB, 32 * WPB, 0, stream>>>(qm, km, vt, ao);
  int waves_o = (B_ * S_ / 16) * (D_ / 64);            // 4096
  oproj_kernel<<<waves_o / 4, 128, 0, stream>>>(ao, wot, bo, out);
}
